// ObjectModel_40372692582531
// MI455X (gfx1250) — matrix operand order repaired and checked by reading, not compiled, not run on hardware
//
#include <hip/hip_runtime.h>
#include <hip/hip_bf16.h>

// ---------------------------------------------------------------------------
// Shapes (fixed by the reference): B=16, L1=L2=2048, H=768.
// Workspace layout (~145 MB):
//   [0, 48MB)      x1wb : bf16 [B][L1][H]   projected queries (GEMM1 out)
//   [+48MB)        x2b  : bf16 [B][L2][H]   x2, key-major   (scores B operand)
//   [+96MB)        x2bT : bf16 [B][H][L2]   x2, H-major     (P@V    B operand)
//   [+144MB)       wt   : bf16 [H][H]       w[0:768,:] transposed (N-major)
//   [.., +48KB)    pool : u32  [B][H]       order-encoded running max
// ---------------------------------------------------------------------------

typedef __attribute__((ext_vector_type(16))) __bf16 bf16x16;
typedef __attribute__((ext_vector_type(8)))  __bf16 bf16x8;
typedef __attribute__((ext_vector_type(4)))  __bf16 bf16x4;
typedef __attribute__((ext_vector_type(8)))  float  f32x8;
typedef __attribute__((ext_vector_type(4)))  float  f32x4;

#define B_   16
#define L1_  2048
#define L2_  2048
#define H_   768
#define CH_  32    // keys per attention chunk
#define TSTR 72    // cvt-transpose LDS tile stride (bf16 elems)

__device__ __forceinline__ f32x8 wmma_bf16(bf16x16 a, bf16x16 b, f32x8 c) {
  // D = A(16x32 bf16) * B(32x16 bf16) + C(16x16 f32)
  return __builtin_amdgcn_wmma_f32_16x16x32_bf16(false, a, false, b, (short)0, c,
                                                 false, false);
}

__device__ __forceinline__ bf16x16 join8(bf16x8 lo, bf16x8 hi) {
  return __builtin_shufflevector(lo, hi, 0, 1, 2, 3, 4, 5, 6, 7, 8, 9, 10, 11, 12, 13, 14, 15);
}

// Order-preserving f32 <-> u32 encoding so atomicMax(u32) == float max.
__device__ __forceinline__ unsigned enc_ord(float f) {
  unsigned b = __float_as_uint(f);
  return (b & 0x80000000u) ? ~b : (b | 0x80000000u);
}
__device__ __forceinline__ float dec_ord(unsigned k) {
  unsigned b = (k & 0x80000000u) ? (k ^ 0x80000000u) : ~k;
  return __uint_as_float(b);
}

// ---------------------------------------------------------------------------
// prep: wt[n][k] = bf16(w[k][n]); pool init; inverted masks -> d_out
// ---------------------------------------------------------------------------
__global__ __launch_bounds__(256) void prep_kernel(const float* __restrict__ w,
                                                   const int* __restrict__ x1_mask,
                                                   const int* __restrict__ x2_mask,
                                                   __bf16* __restrict__ wt,
                                                   unsigned* __restrict__ pool,
                                                   float* __restrict__ out) {
  const int i = blockIdx.x * 256 + threadIdx.x;
  if (i < H_ * H_) {
    const int n = i / H_, k = i % H_;
    wt[(size_t)n * H_ + k] = (__bf16)w[(size_t)k * H_ + n];
  }
  if (i < B_ * H_) pool[i] = enc_ord(-100000.0f);
  if (i < B_ * L1_) out[16 + i] = (x1_mask[i] == 0) ? 1.0f : 0.0f;
  if (i < B_ * L2_) out[16 + B_ * L1_ + i] = (x2_mask[i] == 0) ? 1.0f : 0.0f;
}

// ---------------------------------------------------------------------------
// cvt_x2: x2 (f32) -> x2b (bf16 key-major) and x2bT (bf16 H-major) via LDS
// 64x64 tile transpose. Coalesced reads and coalesced writes both ways.
// grid = B * (L2/64) * (H/64) = 6144 blocks.
// ---------------------------------------------------------------------------
__global__ __launch_bounds__(256) void cvt_x2_kernel(const float* __restrict__ x2,
                                                     __bf16* __restrict__ x2b,
                                                     __bf16* __restrict__ x2bT) {
  __shared__ __bf16 tile[64 * TSTR];
  const int blk = blockIdx.x;
  const int hb = blk % (H_ / 64);
  const int lb = (blk / (H_ / 64)) % (L2_ / 64);
  const int b  = blk / ((H_ / 64) * (L2_ / 64));
  const int h0 = hb * 64, l0 = lb * 64;
  const int tid = threadIdx.x;

  // phase 1: load 64x64 f32 tile, cvt, write x2b directly + LDS tile
  {
    const int r0 = tid >> 4;         // 0..15
    const int c  = (tid & 15) * 4;   // 0..60
#pragma unroll
    for (int i = 0; i < 4; ++i) {
      const int r = r0 + 16 * i;
      const size_t row = (size_t)(b * L2_ + l0 + r);
      f32x4 v = *(const f32x4*)(x2 + row * H_ + h0 + c);
      bf16x4 bv;
#pragma unroll
      for (int e = 0; e < 4; ++e) bv[e] = (__bf16)v[e];
      *(bf16x4*)(x2b + row * H_ + h0 + c) = bv;
      *(bf16x4*)(tile + r * TSTR + c) = bv;
    }
  }
  __syncthreads();
  // phase 2: write x2bT rows (H-major), gathering LDS columns
#pragma unroll
  for (int i = 0; i < 2; ++i) {
    const int j = tid + 256 * i;     // 0..511
    const int hh = j >> 3;           // 0..63  (local H index)
    const int seg = j & 7;           // 8-elem segment along L
    bf16x8 ov;
#pragma unroll
    for (int e = 0; e < 8; ++e) ov[e] = tile[(seg * 8 + e) * TSTR + hh];
    *(bf16x8*)(x2bT + ((size_t)(b * H_ + h0 + hh)) * L2_ + l0 + seg * 8) = ov;
  }
}

// ---------------------------------------------------------------------------
// gemm1: x1wb = bf16( concat(x1, y) @ w )    [B*L1, 768] = [.,769]x[769,768]
// 8 waves: wave w -> M-tile (w&1), 192-wide N slice ((w>>1)*192). K loop 768.
// ---------------------------------------------------------------------------
__global__ __launch_bounds__(256) void gemm1_kernel(const float* __restrict__ x1,
                                                    const float* __restrict__ y,
                                                    const float* __restrict__ w,
                                                    const __bf16* __restrict__ wt,
                                                    __bf16* __restrict__ x1wb) {
  const int blk = blockIdx.x;          // B_ * L1_/32 = 1024
  const int b = blk >> 6;
  const int q0 = (blk & 63) * 32;
  const int tid = threadIdx.x;
  const int wv = tid >> 5, lane = tid & 31, l15 = lane & 15, h = lane >> 4;
  const int mt = wv & 1;
  const int nbase = (wv >> 1) * 192;
  const int row = q0 + mt * 16 + l15;
  const float* arow = x1 + ((size_t)b * L1_ + row) * H_;

  f32x8 acc[12];
#pragma unroll
  for (int j = 0; j < 12; ++j) acc[j] = (f32x8)(0.0f);

  for (int kb = 0; kb < H_; kb += 32) {
    // A operand: lane row, K = {kb+16h..+15}  (16x32 bf16 layout)
    f32x4 a0 = *(const f32x4*)(arow + kb + 16 * h);
    f32x4 a1 = *(const f32x4*)(arow + kb + 16 * h + 4);
    f32x4 a2 = *(const f32x4*)(arow + kb + 16 * h + 8);
    f32x4 a3 = *(const f32x4*)(arow + kb + 16 * h + 12);
    bf16x16 av;
#pragma unroll
    for (int i = 0; i < 4; ++i) {
      av[i]      = (__bf16)a0[i];
      av[4 + i]  = (__bf16)a1[i];
      av[8 + i]  = (__bf16)a2[i];
      av[12 + i] = (__bf16)a3[i];
    }
#pragma unroll
    for (int j = 0; j < 12; ++j) {
      const __bf16* bp = wt + (size_t)(nbase + 16 * j + l15) * H_ + kb + 16 * h;
      bf16x16 bv = *(const bf16x16*)bp;   // 32B contiguous, K-major column of w
      acc[j] = wmma_bf16(av, bv, acc[j]);
    }
  }

  // epilogue: add rank-1 y * w[768,:] term, convert, store bf16
  const float myY = y[(size_t)b * L1_ + row];
#pragma unroll
  for (int j = 0; j < 12; ++j) {
    const int col = nbase + 16 * j + l15;
    const float wl = w[(size_t)H_ * H_ + col];  // w[768][col]
#pragma unroll
    for (int r = 0; r < 8; ++r) {
      const float yv = __shfl(myY, r + 8 * h, 32);
      const int orow = q0 + mt * 16 + r + 8 * h;
      const float val = acc[j][r] + yv * wl;
      x1wb[((size_t)b * L1_ + orow) * H_ + col] = (__bf16)val;
    }
  }
}

// ---------------------------------------------------------------------------
// attn: fused masked-softmax attention + masked max-pool (flash style).
// WG = 8 waves / 32 queries. Wave w owns H-slice [w*96, +96) for both the
// split-K score partial and the P@V accumulation. All WMMA operands are
// contiguous 32B global loads out of L2 (x2 pre-converted to both layouts);
// LDS is only the deterministic cross-wave score reduction.
// ---------------------------------------------------------------------------
__global__ __launch_bounds__(256) void attn_kernel(const __bf16* __restrict__ x1wb,
                                                   const __bf16* __restrict__ x2b,
                                                   const __bf16* __restrict__ x2bT,
                                                   const int* __restrict__ x2_mask,
                                                   const int* __restrict__ x1_mask,
                                                   unsigned* __restrict__ pool) {
  __shared__ float Spart[8 * 32 * 32];  // per-wave K-split score partials
  __shared__ float Sbuf[32 * 32];       // reduced + masked scores
  __shared__ int   kmask[CH_];

  const int blk = blockIdx.x;          // 1024
  const int b = blk >> 6;
  const int q0 = (blk & 63) * 32;
  const int tid = threadIdx.x;
  const int wv = tid >> 5, lane = tid & 31, l15 = lane & 15, h = lane >> 4;
  const int hs = wv * 96;

  // Persistent Q A-operands for this wave's K slice: [mtile][kstep]
  bf16x16 Qa[2][3];
#pragma unroll
  for (int mt = 0; mt < 2; ++mt) {
    const int row = q0 + mt * 16 + l15;
    const __bf16* qrow = x1wb + ((size_t)b * L1_ + row) * H_;
#pragma unroll
    for (int ks = 0; ks < 3; ++ks) {
      const __bf16* p = qrow + hs + 32 * ks;
      bf16x8 lo = *(const bf16x8*)(p + 16 * h);
      bf16x8 hi = *(const bf16x8*)(p + 16 * h + 8);
      Qa[mt][ks] = join8(lo, hi);
    }
  }

  f32x8 acc[2][6];
#pragma unroll
  for (int mt = 0; mt < 2; ++mt)
#pragma unroll
    for (int ht = 0; ht < 6; ++ht) acc[mt][ht] = (f32x8)(0.0f);

  float mrun[2] = {-3.0e38f, -3.0e38f};
  float lrun[2] = {0.0f, 0.0f};

  for (int n0 = 0; n0 < L2_; n0 += CH_) {
    __syncthreads();  // protect Spart/Sbuf/kmask reuse
    if (tid < CH_) kmask[tid] = x2_mask[b * L2_ + n0 + tid];
    if (tid < CH_ && n0 + CH_ < L2_)
      __builtin_prefetch(x2b + ((size_t)(b * L2_ + n0 + CH_ + tid)) * H_ + hs, 0, 1);

    // ---- partial scores over this wave's 96-wide K slice (B loaded once) ----
    f32x8 sc[2][2];  // [mt][nt]
    sc[0][0] = (f32x8)(0.0f); sc[0][1] = (f32x8)(0.0f);
    sc[1][0] = (f32x8)(0.0f); sc[1][1] = (f32x8)(0.0f);
#pragma unroll
    for (int ks = 0; ks < 3; ++ks) {
#pragma unroll
      for (int nt = 0; nt < 2; ++nt) {
        const __bf16* bp =
            x2b + ((size_t)(b * L2_ + n0 + nt * 16 + l15)) * H_ + hs + 32 * ks + 16 * h;
        bf16x16 bv = *(const bf16x16*)bp;  // 32B contiguous key row
        sc[0][nt] = wmma_bf16(Qa[0][ks], bv, sc[0][nt]);
        sc[1][nt] = wmma_bf16(Qa[1][ks], bv, sc[1][nt]);
      }
    }
#pragma unroll
    for (int mt = 0; mt < 2; ++mt)
#pragma unroll
      for (int nt = 0; nt < 2; ++nt)
#pragma unroll
        for (int r = 0; r < 8; ++r)
          Spart[wv * 1024 + (mt * 16 + r + 8 * h) * 32 + nt * 16 + l15] = sc[mt][nt][r];
    __syncthreads();

    // ---- deterministic cross-wave reduce + masked fill (-1e-5) ----
#pragma unroll
    for (int j = 0; j < 4; ++j) {
      const int idx = tid + 256 * j;  // [0,1024)
      float s = 0.0f;
#pragma unroll
      for (int p8 = 0; p8 < 8; ++p8) s += Spart[p8 * 1024 + idx];
      if (kmask[idx & 31] == 0) s = -1e-5f;
      Sbuf[idx] = s;
    }
    __syncthreads();

    // ---- online softmax: stats, P pack, acc rescale ----
    bf16x16 P[2];
#pragma unroll
    for (int mt = 0; mt < 2; ++mt) {
      const float* srow = Sbuf + (mt * 16 + l15) * 32;
      float sv[16];
#pragma unroll
      for (int i = 0; i < 8; ++i) {
        sv[i] = srow[16 * h + i];          // keys {0..7}+16h
        sv[8 + i] = srow[16 * h + 8 + i];  // keys {8..15}+16h
      }
      float mx = sv[0];
#pragma unroll
      for (int i = 1; i < 16; ++i) mx = fmaxf(mx, sv[i]);
      mx = fmaxf(mx, __shfl_xor(mx, 16, 32));
      const float mnew = fmaxf(mrun[mt], mx);
      const float scale = __expf(mrun[mt] - mnew);
      float ssum = 0.0f;
#pragma unroll
      for (int i = 0; i < 16; ++i) {
        const float e = __expf(sv[i] - mnew);
        P[mt][i] = (__bf16)e;
        ssum += e;
      }
      ssum += __shfl_xor(ssum, 16, 32);
      lrun[mt] = lrun[mt] * scale + ssum;
      mrun[mt] = mnew;
#pragma unroll
      for (int r = 0; r < 8; ++r) {
        const float scr = __shfl(scale, r + 8 * h, 32);
#pragma unroll
        for (int ht = 0; ht < 6; ++ht) acc[mt][ht][r] *= scr;
      }
    }

    // ---- P @ V over this wave's 96 output columns (B from x2bT, 32B loads) ----
#pragma unroll
    for (int ht = 0; ht < 6; ++ht) {
      const __bf16* vp =
          x2bT + ((size_t)(b * H_ + hs + 16 * ht + l15)) * L2_ + n0 + 16 * h;
      bf16x16 bv = *(const bf16x16*)vp;  // 32B contiguous keys for this column
      acc[0][ht] = wmma_bf16(P[0], bv, acc[0][ht]);
      acc[1][ht] = wmma_bf16(P[1], bv, acc[1][ht]);
    }
  }

  // ---- epilogue: normalize, x1-masked max-pool via ordered-u32 atomicMax ----
  int qm[2];
  qm[0] = x1_mask[b * L1_ + q0 + l15];
  qm[1] = x1_mask[b * L1_ + q0 + 16 + l15];
  float linv[2] = {1.0f / lrun[0], 1.0f / lrun[1]};
  unsigned* poolb = pool + (size_t)b * H_;
#pragma unroll
  for (int ht = 0; ht < 6; ++ht) {
    float mx = -3.0e38f;
#pragma unroll
    for (int mt = 0; mt < 2; ++mt) {
#pragma unroll
      for (int r = 0; r < 8; ++r) {
        const float li = __shfl(linv[mt], r + 8 * h, 32);
        const int qq = __shfl(qm[mt], r + 8 * h, 32);
        const float v = acc[mt][ht][r] * li;
        if (qq != 0) mx = fmaxf(mx, v);  // masked rows == pool init (-1e5)
      }
    }
    mx = fmaxf(mx, __shfl_xor(mx, 16, 32));
    if (lane < 16) atomicMax(poolb + hs + 16 * ht + l15, enc_ord(mx));
  }
}

// ---------------------------------------------------------------------------
// final: o[b] = sigmoid( [x1_h, x2_h, pool] . lin_w + lin_b )
// ---------------------------------------------------------------------------
__global__ __launch_bounds__(256) void final_kernel(const float* __restrict__ x1h,
                                                    const float* __restrict__ x2h,
                                                    const float* __restrict__ lw,
                                                    const float* __restrict__ lb,
                                                    const unsigned* __restrict__ pool,
                                                    float* __restrict__ out) {
  __shared__ float red[256];
  const int b = blockIdx.x, tid = threadIdx.x;
  float p = 0.0f;
  for (int i = tid; i < H_; i += 256) {
    p += x1h[b * H_ + i] * lw[i];
    p += x2h[b * H_ + i] * lw[H_ + i];
    p += dec_ord(pool[b * H_ + i]) * lw[2 * H_ + i];
  }
  red[tid] = p;
  __syncthreads();
  for (int s = 128; s > 0; s >>= 1) {
    if (tid < s) red[tid] += red[tid + s];
    __syncthreads();
  }
  if (tid == 0) out[b] = 1.0f / (1.0f + __expf(-(red[0] + lb[0])));
}

// ---------------------------------------------------------------------------
extern "C" void kernel_launch(void* const* d_in, const int* in_sizes, int n_in,
                              void* d_out, int out_size, void* d_ws, size_t ws_size,
                              hipStream_t stream) {
  const float* x1  = (const float*)d_in[0];
  const float* x1h = (const float*)d_in[1];
  const int*   x1m = (const int*)d_in[2];
  const float* y   = (const float*)d_in[3];
  const float* x2  = (const float*)d_in[4];
  const float* x2h = (const float*)d_in[5];
  const int*   x2m = (const int*)d_in[6];
  const float* w   = (const float*)d_in[7];
  const float* lw  = (const float*)d_in[8];
  const float* lb  = (const float*)d_in[9];
  float* out = (float*)d_out;

  char* ws = (char*)d_ws;
  const size_t sz_act = (size_t)B_ * L1_ * H_ * 2;  // 48 MB (L1_ == L2_)
  __bf16* x1wb = (__bf16*)ws;
  __bf16* x2b  = (__bf16*)(ws + sz_act);
  __bf16* x2bT = (__bf16*)(ws + 2 * sz_act);
  __bf16* wt   = (__bf16*)(ws + 3 * sz_act);
  unsigned* pool = (unsigned*)(ws + 3 * sz_act + (size_t)H_ * H_ * 2);

  prep_kernel<<<(H_ * H_ + 255) / 256, 256, 0, stream>>>(w, x1m, x2m, wt, pool, out);
  cvt_x2_kernel<<<B_ * (L2_ / 64) * (H_ / 64), 256, 0, stream>>>(x2, x2b, x2bT);
  gemm1_kernel<<<B_ * L1_ / 32, 256, 0, stream>>>(x1, y, w, wt, x1wb);
  attn_kernel<<<B_ * L1_ / 32, 256, 0, stream>>>(x1wb, x2b, x2bT, x2m, x1m, pool);
  final_kernel<<<B_, 256, 0, stream>>>(x1h, x2h, lw, lb, pool, out);
}
